// MultiHeadAttention_86595130622548
// MI455X (gfx1250) — matrix operand order repaired and checked by reading, not compiled, not run on hardware
//
#include <hip/hip_runtime.h>

// MHA forward for B=2, S=2048, H=1024, NH=16, HD=64 on gfx1250 (MI455X).
// f16 WMMA (v_wmma_f32_16x16x32_f16) with f32 accumulation, flash-style
// single-pass attention. K/V tiles staged in LDS by the Tensor Data Mover
// (tensor_load_to_lds, TENSORcnt), double-buffered, shared by 8 waves.

typedef _Float16 half_t;
typedef __attribute__((ext_vector_type(16))) _Float16 v16h;
typedef __attribute__((ext_vector_type(8)))  float    v8f;
typedef __attribute__((ext_vector_type(4)))  unsigned int u32x4;
typedef __attribute__((ext_vector_type(8)))  int i32x8;
typedef __attribute__((ext_vector_type(4)))  int i32x4;

#define NBATCH 2
#define SEQ    2048
#define HID    1024
#define NHEAD  16
#define HD     64
#define MROWS  (NBATCH * SEQ)   // 4096

union HalfFrag {
    v16h  h;
    uint4 u[2];
    half_t e[16];
};

// -------------------------------------------------------------------------
// mask -> additive bias: 0 where mask==1, -1e10 where mask==0
__global__ __launch_bounds__(256) void mha_maskbias_kernel(
    const int* __restrict__ mask, float* __restrict__ mb)
{
    int i = blockIdx.x * blockDim.x + threadIdx.x;
    if (i < NBATCH * SEQ) mb[i] = (mask[i] == 0) ? -1e10f : 0.0f;
}

// -------------------------------------------------------------------------
// Projection GEMM: out = X @ W^T + b, f32 in, f16 out.
// Each wave computes one 16x16 output tile; K-loop of 32 WMMAs.
// vt_layout==0: out[(b*NH+h)*SEQ + s][d]   (Q, K layout)
// vt_layout==1: out[(b*NH+h)*HD  + d][s]   (V transposed for PV B-fragments)
__global__ __launch_bounds__(256) void mha_proj_kernel(
    const float* __restrict__ X, const float* __restrict__ W,
    const float* __restrict__ bias, half_t* __restrict__ out, int vt_layout)
{
    const int wave = threadIdx.x >> 5;
    const int lane = threadIdx.x & 31;
    const int ln   = lane & 15;
    const int hi   = lane >> 4;                 // 0: lanes 0-15, 1: lanes 16-31

    const int tile  = blockIdx.x * 8 + wave;    // 256*64 = 16384 tiles
    const int mtile = tile >> 6;                // N tiles per row = 1024/16 = 64
    const int ntile = tile & 63;
    const int mbase = mtile * 16;
    const int nbase = ntile * 16;

    const float* xrow = X + (long)(mbase + ln) * HID;  // A: row per lane
    const float* wrow = W + (long)(nbase + ln) * HID;  // B: out-feature per lane

    v8f acc = {};
    for (int kb = 0; kb < HID; kb += 32) {
        HalfFrag a, b;
        // A-layout (16x32 f16): lane<16 holds k {0..7,16..23}, lane>=16 {8..15,24..31}
        #pragma unroll
        for (int i = 0; i < 8; ++i) a.e[i]     = (half_t)xrow[kb + hi * 8 + i];
        #pragma unroll
        for (int i = 0; i < 8; ++i) a.e[8 + i] = (half_t)xrow[kb + 16 + hi * 8 + i];
        // B-layout (32x16 f16): lane<16 holds k {0..7,16..23}, lane>=16 {8..15,24..31}
        #pragma unroll
        for (int i = 0; i < 16; ++i) b.e[i]    = (half_t)wrow[kb + (i >> 3) * 16 + hi * 8 + (i & 7)];

        acc = __builtin_amdgcn_wmma_f32_16x16x32_f16(
            false, a.h, false, b.h, (short)0, acc, false, false);
    }

    const int   n  = nbase + ln;
    const float bn = bias[n];
    const int   h  = n >> 6;           // head
    const int   d  = n & 63;           // dim within head
    #pragma unroll
    for (int r = 0; r < 8; ++r) {
        const int m  = mbase + r + hi * 8;   // C/D layout: M = r (+8 for hi lanes)
        const int bb = m >> 11;              // batch
        const int s  = m & 2047;             // seq
        const float v = acc[r] + bn;
        long idx;
        if (vt_layout) idx = ((long)(bb * NHEAD + h) * HD  + d) * SEQ + s;
        else           idx = ((long)(bb * NHEAD + h) * SEQ + s) * HD  + d;
        out[idx] = (half_t)v;
    }
}

// -------------------------------------------------------------------------
// Issue one 2-D TDM tile load: global (f16, row stride `stride0` elements,
// tile tile_d0 x tile_d1) -> compact LDS tile at byte offset lds_off.
// D# layout per CDNA5 ISA ch.8 (group0: flags/lds/global/type, group1: dims).
// This toolchain (clang-23 / therock-10.0) uses the 6-arg builtin form:
// (u32x4 g0, i32x8 g1, i32x4 g2, i32x4 g3, i32x8 extra, i32 cpol).
__device__ __forceinline__ void tdm_load_2d(
    const half_t* gptr, unsigned lds_off,
    unsigned tile_d0, unsigned tile_d1,
    unsigned tensor_d0, unsigned tensor_d1, unsigned stride0)
{
    const unsigned long long ga = (unsigned long long)(uintptr_t)gptr;
    u32x4 g0;
    g0[0] = 1u;                                   // count=1, user descriptor
    g0[1] = lds_off;                              // lds_addr (bytes)
    g0[2] = (unsigned)ga;                         // global_addr[31:0]
    g0[3] = (unsigned)((ga >> 32) & 0x01FFFFFFu)  // global_addr[56:32]
          | (2u << 30);                           // type = 2 ("image")
    i32x8 g1;
    g1[0] = (int)(1u << 16);                      // data_size = 1 -> 2 bytes
    g1[1] = (int)((tensor_d0 & 0xFFFFu) << 16);   // tensor_dim0[15:0]
    g1[2] = (int)((tensor_d0 >> 16) | ((tensor_d1 & 0xFFFFu) << 16));
    g1[3] = (int)((tensor_d1 >> 16) | (tile_d0 << 16));   // tile_dim0
    g1[4] = (int)(tile_d1 & 0xFFFFu);             // tile_dim1 (tile_dim2 = 0)
    g1[5] = (int)stride0;                         // tensor_dim0_stride[31:0]
    g1[6] = 0;                                    // stride0 hi | stride1 lo
    g1[7] = 0;
    i32x4 z4 = {0, 0, 0, 0};                      // 2-D tensor: groups 2/3 unused
    i32x8 z8 = {0, 0, 0, 0, 0, 0, 0, 0};
    __builtin_amdgcn_tensor_load_to_lds(g0, g1, z4, z4, z8, 0);
}

// -------------------------------------------------------------------------
// Flash attention: block = one (b, head) x 8 query tiles (one per wave).
// 32 keys/iteration: TDM double-buffers K (32x64) and V^T (64x32) tiles in
// LDS; each wave does 4 WMMAs for QK^T, online softmax, 4 WMMAs for PV.
__global__ __launch_bounds__(256) void mha_flash_kernel(
    const half_t* __restrict__ Qh, const half_t* __restrict__ Kh,
    const half_t* __restrict__ Vth, const float* __restrict__ maskb,
    float* __restrict__ out)
{
    // [0,8KB): K tiles x2 | [8KB,16KB): V tiles x2 | [16KB,24KB): P scratch x8
    __shared__ __align__(16) half_t smem[2 * 2048 + 2 * 2048 + 8 * 512];

    const int wave = threadIdx.x >> 5;
    const int lane = threadIdx.x & 31;
    const int ln   = lane & 15;
    const int hi   = lane >> 4;

    const int bh     = blockIdx.x >> 4;          // b*NH + h   (32 values)
    const int qgroup = blockIdx.x & 15;          // 16 groups of 8 q-tiles
    const int bb     = bh >> 4;
    const int hh     = bh & 15;
    const int qbase  = (qgroup * 8 + wave) * 16;

    half_t* ldsK = smem;                 // [buf][32 keys][64 d]
    half_t* ldsV = smem + 2 * 2048;      // [buf][64 d][32 s]
    half_t* ldsP = smem + 4 * 2048 + wave * 512;
    const unsigned smem_base = (unsigned)(uintptr_t)(void*)smem;
    const unsigned offK = smem_base;
    const unsigned offV = smem_base + 2 * 2048 * 2;

    const half_t* kp = Kh  + (long)bh * SEQ * HD;
    const half_t* vp = Vth + (long)bh * HD  * SEQ;

    // Q fragments (A-layout, 16x32 each), resident for the whole pass
    const half_t* qrow = Qh + ((long)bh * SEQ + qbase + ln) * HD;
    HalfFrag qa[2];
    #pragma unroll
    for (int f = 0; f < 2; ++f) {
        qa[f].u[0] = *(const uint4*)(qrow + f * 32 + hi * 8);
        qa[f].u[1] = *(const uint4*)(qrow + f * 32 + 16 + hi * 8);
    }

    float m_run[8], l_run[8], alpha[8];
    v8f o[4] = {};
    #pragma unroll
    for (int r = 0; r < 8; ++r) { m_run[r] = -1e30f; l_run[r] = 0.0f; }
    const float* mbp = maskb + bb * SEQ;

    // Prologue: DMA first K/V tile into buffer 0 (wave 0 drives the TDM)
    if (wave == 0) {
        tdm_load_2d(kp, offK, /*tile*/ HD, 32, /*tensor*/ HD, SEQ, HD);
        tdm_load_2d(vp, offV, /*tile*/ 32, HD, /*tensor*/ SEQ, HD, SEQ);
    }

    for (int j = 0; j < SEQ; j += 32) {
        const int cur = (j >> 5) & 1;
        if (wave == 0) {
            if (j + 32 < SEQ) {   // overlap next tile DMA with current compute
                const int nxt = cur ^ 1;
                tdm_load_2d(kp + (long)(j + 32) * HD, offK + nxt * 4096,
                            HD, 32, HD, SEQ, HD);
                tdm_load_2d(vp + (j + 32), offV + nxt * 4096,
                            32, HD, SEQ, HD, SEQ);
                __builtin_amdgcn_s_wait_tensorcnt(2);  // current buffer done
            } else {
                __builtin_amdgcn_s_wait_tensorcnt(0);
            }
        }
        __syncthreads();   // publish TDM-written LDS tile to all waves

        const half_t* kt = ldsK + cur * 2048;
        const half_t* vt = ldsV + cur * 2048;

        // ---- scores: two 16x16 tiles (keys j..j+15, j+16..j+31) ----
        v8f s[2];
        #pragma unroll
        for (int t = 0; t < 2; ++t) {
            const half_t* krow = kt + (t * 16 + ln) * HD;   // lane = key row
            v8f c = {};
            #pragma unroll
            for (int f = 0; f < 2; ++f) {   // split head dim 64 = 2 x K32
                HalfFrag kb;
                kb.u[0] = *(const uint4*)(krow + f * 32 + hi * 8);
                kb.u[1] = *(const uint4*)(krow + f * 32 + 16 + hi * 8);
                c = __builtin_amdgcn_wmma_f32_16x16x32_f16(
                    false, qa[f].h, false, kb.h, (short)0, c, false, false);
            }
            const float bias = mbp[j + t * 16 + ln];   // key-padding bias
            #pragma unroll
            for (int r = 0; r < 8; ++r) c[r] += bias;
            s[t] = c;
        }

        // ---- online softmax (row stats via 16-lane xor reductions) ----
        #pragma unroll
        for (int r = 0; r < 8; ++r) {
            float tv = fmaxf(s[0][r], s[1][r]);
            #pragma unroll
            for (int m = 1; m < 16; m <<= 1) tv = fmaxf(tv, __shfl_xor(tv, m, 16));
            const float mn = fmaxf(m_run[r], tv);
            alpha[r] = __expf(m_run[r] - mn);
            m_run[r] = mn;
            const float p0 = __expf(s[0][r] - mn);
            const float p1 = __expf(s[1][r] - mn);
            s[0][r] = p0; s[1][r] = p1;
            float rs = p0 + p1;
            #pragma unroll
            for (int m = 1; m < 16; m <<= 1) rs += __shfl_xor(rs, m, 16);
            l_run[r] = l_run[r] * alpha[r] + rs;
        }
        #pragma unroll
        for (int d = 0; d < 4; ++d)
            #pragma unroll
            for (int r = 0; r < 8; ++r) o[d][r] *= alpha[r];

        // ---- transpose P (C/D layout -> A layout) through LDS ----
        #pragma unroll
        for (int t = 0; t < 2; ++t)
            #pragma unroll
            for (int r = 0; r < 8; ++r)
                ldsP[(r + hi * 8) * 32 + t * 16 + ln] = (half_t)s[t][r];
        __syncthreads();
        HalfFrag pa;
        pa.u[0] = *(const uint4*)(ldsP + ln * 32 + hi * 8);
        pa.u[1] = *(const uint4*)(ldsP + ln * 32 + 16 + hi * 8);

        // ---- O += P(16x32) @ V(32x64): lane = one head-dim row of V^T ----
        #pragma unroll
        for (int d = 0; d < 4; ++d) {
            HalfFrag vb;
            vb.u[0] = *(const uint4*)(vt + (d * 16 + ln) * 32 + hi * 8);
            vb.u[1] = *(const uint4*)(vt + (d * 16 + ln) * 32 + 16 + hi * 8);
            o[d] = __builtin_amdgcn_wmma_f32_16x16x32_f16(
                false, pa.h, false, vb.h, (short)0, o[d], false, false);
        }
        __syncthreads();   // all reads done before next TDM overwrite / P store
    }

    // ---- epilogue: normalize by l, write [B,S,H] f32 ----
    #pragma unroll
    for (int d = 0; d < 4; ++d) {
        #pragma unroll
        for (int r = 0; r < 8; ++r) {
            const int m = qbase + r + hi * 8;
            const float v = o[d][r] / l_run[r];
            out[((long)(bb * SEQ + m)) * HID + hh * HD + d * 16 + ln] = v;
        }
    }
}

// -------------------------------------------------------------------------
extern "C" void kernel_launch(void* const* d_in, const int* in_sizes, int n_in,
                              void* d_out, int out_size, void* d_ws, size_t ws_size,
                              hipStream_t stream)
{
    const float* query = (const float*)d_in[0];
    const float* key   = (const float*)d_in[1];
    const float* value = (const float*)d_in[2];
    const int*   mask  = (const int*)d_in[3];
    const float* Wq    = (const float*)d_in[4];
    const float* bq    = (const float*)d_in[5];
    const float* Wk    = (const float*)d_in[6];
    const float* bk    = (const float*)d_in[7];
    const float* Wv    = (const float*)d_in[8];
    const float* bv    = (const float*)d_in[9];
    float* out = (float*)d_out;

    // workspace: Qh(8MB) + Kh(8MB) + Vth(8MB) + mask bias(16KB) ~= 24MB
    const long NELEM = (long)MROWS * HID;  // 4194304 halves per tensor
    half_t* Qh  = (half_t*)d_ws;
    half_t* Kh  = Qh + NELEM;
    half_t* Vth = Kh + NELEM;
    float*  mbf = (float*)(Vth + NELEM);

    mha_maskbias_kernel<<<(NBATCH * SEQ) / 256, 256, 0, stream>>>(mask, mbf);

    // 16384 wave-tiles per projection, 8 waves (256 threads) per block
    mha_proj_kernel<<<2048, 256, 0, stream>>>(query, Wq, bq, Qh, 0);
    mha_proj_kernel<<<2048, 256, 0, stream>>>(key,   Wk, bk, Kh, 0);
    mha_proj_kernel<<<2048, 256, 0, stream>>>(value, Wv, bv, Vth, 1);

    // 512 blocks: 32 (b,h) x 16 q-groups; 8 waves per block share K/V tiles
    mha_flash_kernel<<<512, 256, 0, stream>>>(Qh, Kh, Vth, mbf, out);
}
